// UserInvariantInterestModel_39298950758629
// MI455X (gfx1250) — compile-verified, matrix operand order checked by reading, not run on hardware
//
#include <hip/hip_runtime.h>
#include <hip/hip_bf16.h>

typedef __attribute__((ext_vector_type(16))) __bf16 v16bf;
typedef __attribute__((ext_vector_type(8)))  float  v8f;

#define B_    256
#define H_    200
#define HP    208      // H padded to multiple of 16
#define T_    8
#define PCA_  128
#define DL    64
#define FH    140      // x_history row stride
#define FT    138      // x_target row stride

// ---------------------------------------------------------------------------
// Kernel A: lab_h[b,h,:] = w1 @ [c+s, relu(sent@sw^T+sb), type, read, scr] + b1
// Writes fp32 and bf16 copies, padded (zeros) to HP rows per batch.
// One 64-thread block per (b, h) row.
// ---------------------------------------------------------------------------
__global__ void feat_hist_kernel(const float* __restrict__ xh,
                                 const float* __restrict__ cat_emb,
                                 const float* __restrict__ sent_w,
                                 const float* __restrict__ sent_b,
                                 const float* __restrict__ type_emb,
                                 const float* __restrict__ w1_w,
                                 const float* __restrict__ w1_b,
                                 float*  __restrict__ lab_hf,
                                 __bf16* __restrict__ lab_hbf) {
    int bid = blockIdx.x;
    int b = bid / HP, h = bid % HP;
    int o = threadIdx.x;                 // 0..63
    size_t orow = ((size_t)b * HP + h) * DL;
    if (h >= H_) {                       // zero-pad rows (block-uniform branch)
        lab_hf[orow + o]  = 0.0f;
        lab_hbf[orow + o] = (__bf16)0.0f;
        return;
    }
    __shared__ float v[66];
    const float* row = xh + ((size_t)b * H_ + h) * FH;
    if (o < 32) {
        int ci = (int)row[PCA_];
        float c = cat_emb[ci * 32 + o];
        float s = 0.0f;
        #pragma unroll
        for (int k = 0; k < 5; ++k) {
            int si = (int)row[PCA_ + 1 + k];
            s += cat_emb[si * 32 + o];
        }
        v[o] = c + s * 0.2f;
    } else if (o < 48) {
        int j = o - 32;
        float a = sent_b[j];
        #pragma unroll
        for (int e = 0; e < 3; ++e) a += sent_w[j * 3 + e] * row[PCA_ + 6 + e];
        v[o] = a > 0.0f ? a : 0.0f;
    } else {
        int j = o - 48;
        int ti = (int)row[PCA_ + 9];
        v[o] = type_emb[ti * 16 + j];
    }
    if (o == 0) { v[64] = row[PCA_ + 10]; v[65] = row[PCA_ + 11]; }
    __syncthreads();
    float acc = w1_b[o];
    const float* wr = w1_w + o * 66;
    #pragma unroll
    for (int f = 0; f < 66; ++f) acc += wr[f] * v[f];
    lab_hf[orow + o]  = acc;
    lab_hbf[orow + o] = (__bf16)acc;
}

// ---------------------------------------------------------------------------
// Kernel A2: padded staging of ti_h (B, HP, 128) as bf16 (WMMA A operand)
// and fp32 (score*hist multiply) -- padding removes all bounds checks from
// the hot attention loop.
// ---------------------------------------------------------------------------
__global__ void tih_pad_kernel(const float* __restrict__ xh,
                               __bf16* __restrict__ ti_hbf,
                               float*  __restrict__ ti_hf) {
    size_t i = (size_t)blockIdx.x * blockDim.x + threadIdx.x;
    const size_t total = (size_t)B_ * HP * PCA_;
    if (i >= total) return;
    int d = (int)(i % PCA_);
    size_t r = i / PCA_;
    int h = (int)(r % HP);
    int b = (int)(r / HP);
    float val = (h < H_) ? xh[((size_t)b * H_ + h) * FH + d] : 0.0f;
    ti_hbf[i] = (__bf16)val;
    ti_hf[i]  = val;
}

// ---------------------------------------------------------------------------
// Kernel B: lab_t (B,T,64) + ec output = [lab_t | ti_t]. One 64-thread block
// per (b,t).
// ---------------------------------------------------------------------------
__global__ void feat_tgt_kernel(const float* __restrict__ xt,
                                const float* __restrict__ cat_emb,
                                const float* __restrict__ sent_w,
                                const float* __restrict__ sent_b,
                                const float* __restrict__ type_emb,
                                float* __restrict__ lab_t,
                                float* __restrict__ ec) {
    int bt = blockIdx.x;                 // 0..B*T-1
    int o = threadIdx.x;                 // 0..63
    __shared__ float v[64];
    const float* row = xt + (size_t)bt * FT;
    if (o < 32) {
        int ci = (int)row[PCA_];
        float c = cat_emb[ci * 32 + o];
        float s = 0.0f;
        #pragma unroll
        for (int k = 0; k < 5; ++k) {
            int si = (int)row[PCA_ + 1 + k];
            s += cat_emb[si * 32 + o];
        }
        v[o] = c + s * 0.2f;
    } else if (o < 48) {
        int j = o - 32;
        float a = sent_b[j];
        #pragma unroll
        for (int e = 0; e < 3; ++e) a += sent_w[j * 3 + e] * row[PCA_ + 6 + e];
        v[o] = a > 0.0f ? a : 0.0f;
    } else {
        int j = o - 48;
        int ti = (int)row[PCA_ + 9];
        v[o] = type_emb[ti * 16 + j];
    }
    __syncthreads();
    lab_t[(size_t)bt * DL + o] = v[o];
    float* e = ec + (size_t)bt * 192;
    e[o]       = v[o];
    e[64 + o]  = row[o];
    e[128 + o] = row[64 + o];
}

// ---------------------------------------------------------------------------
// WMMA column worker: one wave owns output column `nt` of
//   score = histA(bf16, HP x D) @ M^T       (M in LDS, pre-scaled by tgt)
// and loops over m-tiles [mt0, mt1). B fragments + bias are loop-invariant
// (hoisted into VGPRs); the inner loop is: A-fragment clause -> K WMMAs ->
// fast sigmoid (v_exp_f32 + v_rcp_f32, co-executes with the matrix pipe) ->
// multiply by padded fp32 hist -> ds_add_f32 reduction.
// Fragment layouts per CDNA5 ISA 7.12.2 (A: row=lane&15, K split by lane
// half; B operand holds column n with the same K striping; C: VGPR r ->
// M = r + 8*(lane>=16), N = lane&15).
// ---------------------------------------------------------------------------
template <int D>
__device__ __forceinline__ void do_col(int nt, int mt0, int mt1,
                                       const __bf16* __restrict__ histA,
                                       const __bf16* __restrict__ Msh,
                                       const float*  __restrict__ sbias,
                                       const float*  __restrict__ histF,
                                       float* __restrict__ accsh) {
    constexpr int KSTEPS = D / 32;
    int lane = threadIdx.x & 31;
    int half = lane >> 4;
    int lm   = lane & 15;
    int n    = nt * 16 + lm;

    union Frag { v16bf v; uint4 u[2]; };
    Frag bb[KSTEPS];
    const __bf16* brow = Msh + (size_t)n * D;
    #pragma unroll
    for (int kk = 0; kk < KSTEPS; ++kk) {
        bb[kk].u[0] = *reinterpret_cast<const uint4*>(brow + kk * 32 + half * 8);
        bb[kk].u[1] = *reinterpret_cast<const uint4*>(brow + kk * 32 + 16 + half * 8);
    }
    float bias = sbias[n];

    for (int mt = mt0; mt < mt1; ++mt) {
        const __bf16* arow = histA + (size_t)(mt * 16 + lm) * D;
        v8f c = {};
        #pragma unroll
        for (int kk = 0; kk < KSTEPS; ++kk) {
            Frag a;
            a.u[0] = *reinterpret_cast<const uint4*>(arow + kk * 32 + half * 8);
            a.u[1] = *reinterpret_cast<const uint4*>(arow + kk * 32 + 16 + half * 8);
            c = __builtin_amdgcn_wmma_f32_16x16x32_bf16(false, a.v, false,
                                                        bb[kk].v, (short)0, c,
                                                        false, false);
        }
        const float* hcol = histF + (size_t)(mt * 16 + half * 8) * D + n;
        float partial = 0.0f;
        #pragma unroll
        for (int r = 0; r < 8; ++r) {
            // fast sigmoid: v_exp_f32 + v_rcp_f32 (no IEEE divide chain)
            float e  = __expf(-(c[r] + bias));
            float sc = __builtin_amdgcn_rcpf(1.0f + e);
            partial += sc * hcol[(size_t)r * D];   // padded: always in-bounds
        }
        atomicAdd(&accsh[n], partial);             // ds_add_f32
    }
}

// ---------------------------------------------------------------------------
// Kernel C: both pointwise attentions for one (b,t) per workgroup (8 wave32).
// ---------------------------------------------------------------------------
__global__ void attn_kernel(const float* __restrict__ xt,
                            const float* __restrict__ la_w,
                            const float* __restrict__ la_b,
                            const float* __restrict__ ta_w,
                            const float* __restrict__ ta_b,
                            const float* __restrict__ lab_t,
                            const float* __restrict__ lab_hf,
                            const __bf16* __restrict__ lab_hbf,
                            const float* __restrict__ ti_hf,
                            const __bf16* __restrict__ ti_hbf,
                            float* __restrict__ euH) {
    __shared__ __align__(16) __bf16 Mla[64 * 64];
    __shared__ __align__(16) __bf16 Mta[128 * 128];
    __shared__ float sbias[192];
    __shared__ float acc[192];

    int bt  = blockIdx.x;
    int b   = bt >> 3;
    int tid = threadIdx.x;

    const float* lt = lab_t + (size_t)bt * DL;
    const float* tt = xt + (size_t)bt * FT;
    for (int i = tid; i < 64 * 64; i += 256) {
        int k = i & 63;                  // la_w row-major [n][k]
        Mla[i] = (__bf16)(la_w[i] * lt[k]);
    }
    for (int i = tid; i < 128 * 128; i += 256) {
        int k = i & 127;
        Mta[i] = (__bf16)(ta_w[i] * tt[k]);
    }
    if (tid < 64)  sbias[tid]      = la_b[tid];
    if (tid < 128) sbias[64 + tid] = ta_b[tid];
    if (tid < 192) acc[tid] = 0.0f;
    __syncthreads();

    int wave = tid >> 5;
    // Label attention: 4 columns x 13 m-tiles; split m-range over wave pairs.
    {
        int nt  = wave & 3;
        int mt0 = (wave < 4) ? 0 : 7;
        int mt1 = (wave < 4) ? 7 : 13;
        do_col<64>(nt, mt0, mt1,
                   lab_hbf + (size_t)b * HP * 64, Mla, sbias,
                   lab_hf + (size_t)b * HP * 64, acc);
    }
    // Text-img attention: 8 columns x 13 m-tiles; one column per wave.
    do_col<128>(wave, 0, 13,
                ti_hbf + (size_t)b * HP * 128, Mta, sbias + 64,
                ti_hf + (size_t)b * HP * 128, acc + 64);

    __syncthreads();
    if (tid < 192) euH[(size_t)bt * 192 + tid] = acc[tid];
}

// ---------------------------------------------------------------------------
extern "C" void kernel_launch(void* const* d_in, const int* in_sizes, int n_in,
                              void* d_out, int out_size, void* d_ws, size_t ws_size,
                              hipStream_t stream) {
    const float* xh       = (const float*)d_in[0];   // (256,200,140)
    const float* xt       = (const float*)d_in[1];   // (256,8,138)
    const float* cat_emb  = (const float*)d_in[2];   // (300,32)
    const float* sent_w   = (const float*)d_in[3];   // (16,3)
    const float* sent_b   = (const float*)d_in[4];   // (16,)
    const float* type_emb = (const float*)d_in[5];   // (10,16)
    const float* w1_w     = (const float*)d_in[6];   // (64,66)
    const float* w1_b     = (const float*)d_in[7];   // (64,)
    const float* la_w     = (const float*)d_in[8];   // (64,64)
    const float* la_b     = (const float*)d_in[9];   // (64,)
    const float* ta_w     = (const float*)d_in[10];  // (128,128)
    const float* ta_b     = (const float*)d_in[11];  // (128,)

    float* outp = (float*)d_out;
    float* euH  = outp;                               // (B,T,192)
    float* ec   = outp + (size_t)B_ * T_ * 192;       // (B,T,192)

    char* ws = (char*)d_ws;
    size_t off = 0;
    auto carve = [&](size_t bytes) {
        void* p = ws + off;
        off = (off + bytes + 255) & ~(size_t)255;
        return p;
    };
    __bf16* lab_hbf = (__bf16*)carve((size_t)B_ * HP * 64  * sizeof(__bf16));
    __bf16* ti_hbf  = (__bf16*)carve((size_t)B_ * HP * 128 * sizeof(__bf16));
    float*  lab_hf  = (float*) carve((size_t)B_ * HP * 64  * sizeof(float));
    float*  ti_hf   = (float*) carve((size_t)B_ * HP * 128 * sizeof(float));
    float*  lab_tf  = (float*) carve((size_t)B_ * T_ * 64  * sizeof(float));
    (void)ws_size; (void)in_sizes; (void)n_in; (void)out_size;

    feat_hist_kernel<<<B_ * HP, 64, 0, stream>>>(xh, cat_emb, sent_w, sent_b,
                                                 type_emb, w1_w, w1_b,
                                                 lab_hf, lab_hbf);
    {
        size_t total = (size_t)B_ * HP * PCA_;
        int blocks = (int)((total + 255) / 256);
        tih_pad_kernel<<<blocks, 256, 0, stream>>>(xh, ti_hbf, ti_hf);
    }
    feat_tgt_kernel<<<B_ * T_, 64, 0, stream>>>(xt, cat_emb, sent_w, sent_b,
                                                type_emb, lab_tf, ec);
    attn_kernel<<<B_ * T_, 256, 0, stream>>>(xt, la_w, la_b, ta_w, ta_b,
                                             lab_tf, lab_hf, lab_hbf,
                                             ti_hf, ti_hbf, euH);
}
